// GCN_66692252172820
// MI455X (gfx1250) — compile-verified, matrix operand order checked by reading, not run on hardware
//
#include <hip/hip_runtime.h>
#include <math.h>

#define F_IN 512
#define F_HID 128
#define F_OUT 16

typedef __attribute__((ext_vector_type(16))) __bf16 v16bf;
typedef __attribute__((ext_vector_type(8)))  float  v8f;

union BFrag {
    uint4 u[2];
    v16bf v;
};

__device__ __forceinline__ unsigned int f2bf(float f) {
    union { float f; unsigned int u; } cv; cv.f = f;
    unsigned int u = cv.u;
    // round-to-nearest-even bf16 truncation
    unsigned int r = u + 0x7FFFu + ((u >> 16) & 1u);
    return (r >> 16) & 0xFFFFu;
}

__device__ __forceinline__ uint4 cvt8(float4 a, float4 b) {
    uint4 r;
    r.x = f2bf(a.x) | (f2bf(a.y) << 16);
    r.y = f2bf(a.z) | (f2bf(a.w) << 16);
    r.z = f2bf(b.x) | (f2bf(b.y) << 16);
    r.w = f2bf(b.z) | (f2bf(b.w) << 16);
    return r;
}

// ---------------- degree / norm ----------------

__global__ void k_set_deg(float* deg, int n) {
    int i = blockIdx.x * blockDim.x + threadIdx.x;
    if (i < n) deg[i] = 1.0f;              // self-loop weight
}

__global__ void k_deg_accum(const int* __restrict__ dst,
                            const float* __restrict__ ew,
                            float* deg, int e) {
    int i = blockIdx.x * blockDim.x + threadIdx.x;
    if (i < e) atomicAdd(&deg[dst[i]], ew[i]);
}

__global__ void k_dinv(float* deg, int n) {
    int i = blockIdx.x * blockDim.x + threadIdx.x;
    if (i < n) {
        float d = deg[i];
        deg[i] = (d > 0.0f) ? rsqrtf(d) : 0.0f;
    }
}

// ---------------- weight transpose + bf16 convert ----------------
// Wt1[n][k] = bf16(W1[k][n]); Wt2[n][k] = bf16(W2[k][n])

__global__ void k_prep_weights(const float* __restrict__ W1,
                               const float* __restrict__ W2,
                               unsigned short* __restrict__ Wt1,
                               unsigned short* __restrict__ Wt2) {
    int t = blockIdx.x * blockDim.x + threadIdx.x;
    if (t < F_IN * F_HID) {
        int n = t / F_IN, k = t % F_IN;
        Wt1[t] = (unsigned short)f2bf(W1[k * F_HID + n]);
    }
    int t2 = t - F_IN * F_HID;
    if (t2 >= 0 && t2 < F_HID * F_OUT) {
        int n = t2 / F_HID, k = t2 % F_HID;
        Wt2[t2] = (unsigned short)f2bf(W2[k * F_OUT + n]);
    }
}

// ---------------- GEMM1: h1 = x @ W1  (N x 512) @ (512 x 128) ----------------
// Block: 256 threads = 8 waves. Block handles 16 rows; wave w handles cols [16w,16w+16).
// A tile (16 x 512) staged in LDS as bf16.

__global__ void k_gemm1_wmma(const float* __restrict__ x,
                             const unsigned short* __restrict__ Wt1, // [128][512] bf16
                             float* __restrict__ h1,                 // [N][128]
                             int nrows) {
    __shared__ unsigned short Abuf[16 * F_IN];   // 16 KB bf16
    const int tid = threadIdx.x;
    const int row0 = blockIdx.x * 16;

    // cooperative load + convert A tile
    for (int i = tid; i < 16 * F_IN; i += 256) {
        int m = i / F_IN, k = i % F_IN;
        int gr = row0 + m;
        float v = (gr < nrows) ? x[(size_t)gr * F_IN + k] : 0.0f;
        Abuf[i] = (unsigned short)f2bf(v);
    }
    __syncthreads();

    const int wave = tid >> 5;
    const int lane = tid & 31;
    const int n0 = wave * 16;
    const int mrow = lane & 15;          // A row / B column within tile
    const int khalf = (lane >> 4) << 3;  // 0 or 8

    const unsigned short* brow = Wt1 + (size_t)(n0 + mrow) * F_IN;

    v8f acc = {};
    for (int k = 0; k < F_IN; k += 32) {
        BFrag a, b;
        a.u[0] = *(const uint4*)&Abuf[mrow * F_IN + k + khalf];
        a.u[1] = *(const uint4*)&Abuf[mrow * F_IN + k + 16 + khalf];
        b.u[0] = *(const uint4*)&brow[k + khalf];
        b.u[1] = *(const uint4*)&brow[k + 16 + khalf];
        acc = __builtin_amdgcn_wmma_f32_16x16x32_bf16(
            false, a.v, false, b.v, (short)0, acc, false, false);
    }

    // C layout: VGPR r -> (M = r + 8*(lane>=16), N = lane&15)
    const int col = n0 + (lane & 15);
    const int rbase = row0 + ((lane >> 4) << 3);
#pragma unroll
    for (int r = 0; r < 8; ++r) {
        int row = rbase + r;
        if (row < nrows) h1[(size_t)row * F_HID + col] = acc[r];
    }
}

// ---------------- init agg1 with self-loop term ----------------
// agg1[i][c] = dinv[i]^2 * h1[i][c]     (float4-vectorized: 32 float4 per row)

__global__ void k_init_agg1(const float* __restrict__ h1,
                            const float* __restrict__ dinv,
                            float* __restrict__ agg, int n) {
    int t = blockIdx.x * blockDim.x + threadIdx.x;
    if (t >= n * 32) return;
    int i = t >> 5;
    float s = dinv[i] * dinv[i];
    float4 v = ((const float4*)h1)[t];
    v.x *= s; v.y *= s; v.z *= s; v.w *= s;
    ((float4*)agg)[t] = v;
}

// ---------------- edge aggregation, width 128: one wave per edge ----------------

__global__ void k_edge_agg1(const int* __restrict__ src,
                            const int* __restrict__ dst,
                            const float* __restrict__ ew,
                            const float* __restrict__ dinv,
                            const float* __restrict__ h1,
                            float* __restrict__ agg, int e) {
    int wid = (blockIdx.x * blockDim.x + threadIdx.x) >> 5;
    int lane = threadIdx.x & 31;
    if (wid >= e) return;
    int s = src[wid], d = dst[wid];
    float nrm = dinv[s] * ew[wid] * dinv[d];
    float4 v = ((const float4*)(h1 + (size_t)s * F_HID))[lane];
    float* o = agg + (size_t)d * F_HID + lane * 4;
    atomicAdd(o + 0, v.x * nrm);
    atomicAdd(o + 1, v.y * nrm);
    atomicAdd(o + 2, v.z * nrm);
    atomicAdd(o + 3, v.w * nrm);
}

// ---------------- finalize layer 1: h = relu(agg + b1) ----------------

__global__ void k_finalize1(const float* __restrict__ agg,
                            const float* __restrict__ b1,
                            float* __restrict__ hout, int n) {
    int t = blockIdx.x * blockDim.x + threadIdx.x;
    if (t >= n * 32) return;
    int c4 = (t & 31) * 4;
    float4 v = ((const float4*)agg)[t];
    v.x = fmaxf(v.x + b1[c4 + 0], 0.0f);
    v.y = fmaxf(v.y + b1[c4 + 1], 0.0f);
    v.z = fmaxf(v.z + b1[c4 + 2], 0.0f);
    v.w = fmaxf(v.w + b1[c4 + 3], 0.0f);
    ((float4*)hout)[t] = v;
}

// ---------------- GEMM2: z = h @ W2  (N x 128) @ (128 x 16) ----------------
// Wave per 16-row tile, single 16-col tile, K=128 in 4 WMMA steps. A converted on the fly.

__global__ void k_gemm2_wmma(const float* __restrict__ h,
                             const unsigned short* __restrict__ Wt2, // [16][128] bf16
                             float* __restrict__ z,                  // [N][16]
                             int nrows) {
    const int wave = threadIdx.x >> 5;
    const int lane = threadIdx.x & 31;
    const int m0 = blockIdx.x * 128 + wave * 16;
    if (m0 >= nrows) return;

    const int mrow = lane & 15;
    const int khalf = (lane >> 4) << 3;
    const float* arow = h + (size_t)(m0 + mrow) * F_HID;
    const unsigned short* brow = Wt2 + (size_t)mrow * F_HID;

    v8f acc = {};
#pragma unroll
    for (int k = 0; k < F_HID; k += 32) {
        BFrag a, b;
        float4 a0 = *(const float4*)(arow + k + khalf);
        float4 a1 = *(const float4*)(arow + k + khalf + 4);
        float4 a2 = *(const float4*)(arow + k + 16 + khalf);
        float4 a3 = *(const float4*)(arow + k + 16 + khalf + 4);
        a.u[0] = cvt8(a0, a1);
        a.u[1] = cvt8(a2, a3);
        b.u[0] = *(const uint4*)(brow + k + khalf);
        b.u[1] = *(const uint4*)(brow + k + 16 + khalf);
        acc = __builtin_amdgcn_wmma_f32_16x16x32_bf16(
            false, a.v, false, b.v, (short)0, acc, false, false);
    }

    const int col = lane & 15;
    const int rbase = m0 + ((lane >> 4) << 3);
#pragma unroll
    for (int r = 0; r < 8; ++r) {
        int row = rbase + r;
        if (row < nrows) z[(size_t)row * F_OUT + col] = acc[r];
    }
}

// ---------------- layer-2 aggregation (width 16) ----------------

__global__ void k_init_agg2(const float* __restrict__ z,
                            const float* __restrict__ dinv,
                            float* __restrict__ agg, int n) {
    int t = blockIdx.x * blockDim.x + threadIdx.x;
    if (t >= n * F_OUT) return;
    int i = t >> 4;
    float s = dinv[i] * dinv[i];
    agg[t] = z[t] * s;
}

__global__ void k_edge_agg2(const int* __restrict__ src,
                            const int* __restrict__ dst,
                            const float* __restrict__ ew,
                            const float* __restrict__ dinv,
                            const float* __restrict__ z,
                            float* __restrict__ agg, int e) {
    int t = blockIdx.x * blockDim.x + threadIdx.x;
    if (t >= e * F_OUT) return;
    int ei = t >> 4;
    int c = t & 15;
    int s = src[ei], d = dst[ei];
    float nrm = dinv[s] * ew[ei] * dinv[d];
    atomicAdd(&agg[(size_t)d * F_OUT + c], z[(size_t)s * F_OUT + c] * nrm);
}

// ---------------- bias + log_softmax over 16 features ----------------

__global__ void k_logsoftmax(const float* __restrict__ agg,
                             const float* __restrict__ b2,
                             float* __restrict__ out, int n) {
    int i = blockIdx.x * blockDim.x + threadIdx.x;
    if (i >= n) return;
    float l[F_OUT];
    float m = -INFINITY;
#pragma unroll
    for (int c = 0; c < F_OUT; ++c) {
        l[c] = agg[(size_t)i * F_OUT + c] + b2[c];
        m = fmaxf(m, l[c]);
    }
    float s = 0.0f;
#pragma unroll
    for (int c = 0; c < F_OUT; ++c) s += expf(l[c] - m);
    float lse = m + logf(s);
#pragma unroll
    for (int c = 0; c < F_OUT; ++c) out[(size_t)i * F_OUT + c] = l[c] - lse;
}

// ---------------- orchestration ----------------

extern "C" void kernel_launch(void* const* d_in, const int* in_sizes, int n_in,
                              void* d_out, int out_size, void* d_ws, size_t ws_size,
                              hipStream_t stream) {
    (void)n_in; (void)out_size; (void)ws_size;

    const float* x  = (const float*)d_in[0];
    const int*   ei = (const int*)d_in[1];
    const float* ew = (const float*)d_in[2];
    const float* W1 = (const float*)d_in[3];
    const float* b1 = (const float*)d_in[4];
    const float* W2 = (const float*)d_in[5];
    const float* b2 = (const float*)d_in[6];
    float* out = (float*)d_out;

    const int N = in_sizes[0] / F_IN;
    const int E = in_sizes[2];
    const int* src = ei;
    const int* dst = ei + E;

    // workspace layout (256B-aligned slices)
    char* ws = (char*)d_ws;
    size_t off = 0;
    auto take = [&](size_t bytes) -> char* {
        char* p = ws + off;
        off += (bytes + 255) & ~(size_t)255;
        return p;
    };
    float*          dinv = (float*)take((size_t)N * 4);
    unsigned short* Wt1  = (unsigned short*)take((size_t)F_IN * F_HID * 2);
    unsigned short* Wt2  = (unsigned short*)take((size_t)F_HID * F_OUT * 2);
    float*          bufA = (float*)take((size_t)N * F_HID * 4);  // h1pre, then relu(h)
    float*          bufB = (float*)take((size_t)N * F_HID * 4);  // agg1, then z+agg2
    float* z    = bufB;
    float* agg2 = bufB + (size_t)N * F_OUT;

    const int T = 256;

    // norm coefficients
    k_set_deg<<<(N + T - 1) / T, T, 0, stream>>>(dinv, N);
    k_deg_accum<<<(E + T - 1) / T, T, 0, stream>>>(dst, ew, dinv, E);
    k_dinv<<<(N + T - 1) / T, T, 0, stream>>>(dinv, N);

    // weights to bf16 transposed
    k_prep_weights<<<(F_IN * F_HID + F_HID * F_OUT + T - 1) / T, T, 0, stream>>>(
        W1, W2, Wt1, Wt2);

    // layer 1
    k_gemm1_wmma<<<(N + 15) / 16, T, 0, stream>>>(x, Wt1, bufA, N);
    k_init_agg1<<<((size_t)N * 32 + T - 1) / T, T, 0, stream>>>(bufA, dinv, bufB, N);
    k_edge_agg1<<<(E + 7) / 8, T, 0, stream>>>(src, dst, ew, dinv, bufA, bufB, E);
    k_finalize1<<<((size_t)N * 32 + T - 1) / T, T, 0, stream>>>(bufB, b1, bufA, N);

    // layer 2
    k_gemm2_wmma<<<(N + 127) / 128, T, 0, stream>>>(bufA, Wt2, z, N);
    k_init_agg2<<<((size_t)N * F_OUT + T - 1) / T, T, 0, stream>>>(z, dinv, agg2, N);
    k_edge_agg2<<<((size_t)E * F_OUT + T - 1) / T, T, 0, stream>>>(
        src, dst, ew, dinv, z, agg2, E);

    // output
    k_logsoftmax<<<(N + T - 1) / T, T, 0, stream>>>(agg2, b2, out, N);
}
